// GIPAWideConv_15496242004292
// MI455X (gfx1250) — matrix operand order repaired and checked by reading, not compiled, not run on hardware
//
#include <hip/hip_runtime.h>
#include <cstddef>
#include <cstdint>

// ---------------------------------------------------------------------------
// GIPA wide conv for MI455X (gfx1250, wave32, WMMA)
// Dense projections: v_wmma_f32_16x16x32_bf16, fp32 accumulate.
// A operand staged in LDS pre-swizzled (ds_load_b128), B operand pre-swizzled
// in scratch (global_load_b128). Sparse phase: sortable-uint atomic max +
// f32 atomic adds.
// ---------------------------------------------------------------------------

typedef __bf16 v16bf __attribute__((ext_vector_type(16)));
typedef float  v8f   __attribute__((ext_vector_type(8)));

#define Dk   256
#define HF   128
#define NH   4
#define Fw   32
#define OUTW 32
#define SLOPE 0.2f

// ------------------------- float <-> order-preserving uint ------------------
__device__ __forceinline__ unsigned f2ord(float f) {
    unsigned u = __float_as_uint(f);
    return (u & 0x80000000u) ? ~u : (u | 0x80000000u);
}
__device__ __forceinline__ float ord2f(unsigned u) {
    return (u & 0x80000000u) ? __uint_as_float(u ^ 0x80000000u)
                             : __uint_as_float(~u);
}
#define ORD_NEG_INF 0x007FFFFFu   // f2ord(-inf)

__device__ __forceinline__ float leaky(float x) { return x >= 0.f ? x : SLOPE * x; }

union bfpair { __bf16 b[2]; uint32_t u; };

// ---------------------------------------------------------------------------
// WMMA operand swizzles (ISA 7.12.2, 16-bit operands, wave32):
//  A (16xK): lane = (half<<4)|row ; vgpr pair p: k = kb*32 + (p>>2)*16 + half*8 + (p&3)*2
//  B (Kx16): lane = (half<<4)|col ; vgpr pair p: k = kb*32 + (p>>2)*16 + half*8 + (p&3)*2
//  C/D     : N = lane&15, M = vgpr + 8*(lane>=16)
// Swizzled storage: fragment for (kb, lane) is 16 contiguous bf16 (32 bytes).
// ---------------------------------------------------------------------------

// Wave-level 16x16 tile GEMM over swizzled operands.
template <int K>
__device__ __forceinline__ v8f wmma_tile_sw(const uint32_t* __restrict__ AswLDS,
                                            const __bf16* __restrict__ BswCt) {
    const int lane = threadIdx.x & 31;
    v8f c = {};
#pragma unroll
    for (int kb = 0; kb < (K >> 5); ++kb) {
        v16bf a = *(const v16bf*)(AswLDS + (size_t)(kb * 32 + lane) * 8);
        v16bf b = *(const v16bf*)(BswCt + (size_t)(kb * 32 + lane) * 16);
        c = __builtin_amdgcn_wmma_f32_16x16x32_bf16(false, a, false, b,
                                                    (short)0, c, false, false);
    }
    return c;
}

// Stage a 16-row x K fp32 tile into A-swizzled bf16 LDS (packed u32 pairs).
template <int K>
__device__ __forceinline__ void stage_A_sw(uint32_t* __restrict__ As,
                                           const float* __restrict__ A,
                                           int row0, int Nrows) {
    for (int j = threadIdx.x; j < 8 * K; j += blockDim.x) {   // 16*K/2 pairs
        int p    = j & 7;
        int lane = (j >> 3) & 31;
        int kb   = j >> 8;                                    // 256 pairs / kb
        int r    = lane & 15;
        int k    = kb * 32 + ((p >> 2) << 4) + ((lane >> 4) << 3) + ((p & 3) << 1);
        int row  = row0 + r;
        float x0 = 0.f, x1 = 0.f;
        if (row < Nrows) {
            const float* ar = A + (size_t)row * K;
            x0 = ar[k]; x1 = ar[k + 1];
        }
        bfpair pk; pk.b[0] = (__bf16)x0; pk.b[1] = (__bf16)x1;
        As[j] = pk.u;
    }
}

// ---------------- weight convert + swizzle into B-operand layout ------------
__global__ void cvt_swz_kernel(const float* __restrict__ W,
                               uint32_t* __restrict__ Wsw,  // K*NC/2 u32
                               int K, int NC) {
    int j = blockIdx.x * blockDim.x + threadIdx.x;
    if (j >= (K * NC) / 2) return;
    int pairs_per_ct = K * 8;            // (K/32)*32*8
    int ct   = j / pairs_per_ct;
    int rem  = j % pairs_per_ct;
    int kb   = rem >> 8;
    int lane = (rem >> 3) & 31;
    int p    = rem & 7;
    int col  = ct * 16 + (lane & 15);
    int k    = kb * 32 + ((p >> 2) << 4) + ((lane >> 4) << 3) + ((p & 3) << 1);
    bfpair pk;
    pk.b[0] = (__bf16)W[(size_t)k * NC + col];
    pk.b[1] = (__bf16)W[(size_t)(k + 1) * NC + col];
    Wsw[j] = pk.u;
}

// ------------------------------ init scratch --------------------------------
__global__ void init_kernel(float* __restrict__ h, float* __restrict__ zsum,
                            unsigned* __restrict__ emax, int nH, int n4) {
    int i = blockIdx.x * blockDim.x + threadIdx.x;
    if (i < nH) h[i] = 0.f;
    if (i < n4) { zsum[i] = 0.f; emax[i] = ORD_NEG_INF; }
}

// ------------------- node projections: feat_src_fc, feat_dst_fc -------------
// block = 512 threads (16 waves). Waves 0-7: W_src col tiles; 8-15: W_dst (+b).
__global__ void node_proj_kernel(const float* __restrict__ A,
                                 const __bf16* __restrict__ WsrcSw,
                                 const __bf16* __restrict__ WdstSw,
                                 const float* __restrict__ b_dst,
                                 float* __restrict__ Csrc,
                                 float* __restrict__ Cdst, int Nrows) {
    __shared__ __align__(32) uint32_t As[8 * Dk];   // 16 x 256 bf16, swizzled
    const int row0 = blockIdx.x * 16;
    stage_A_sw<Dk>(As, A, row0, Nrows);
    __syncthreads();
    const int w = threadIdx.x >> 5;
    const bool isDst = w >= 8;
    const int ct = w & 7;
    const __bf16* B = (isDst ? WdstSw : WsrcSw) + (size_t)ct * Dk * 16;
    v8f c = wmma_tile_sw<Dk>(As, B);
    const int lane = threadIdx.x & 31;
    const int hh = lane >> 4, nn = lane & 15;
    const int col0 = ct << 4;
    const float bias = isDst ? b_dst[col0 + nn] : 0.f;
    float* C = isDst ? Cdst : Csrc;
#pragma unroll
    for (int v = 0; v < 8; ++v) {
        int row = row0 + v + (hh << 3);
        if (row < Nrows) C[(size_t)row * HF + col0 + nn] = c[v] + bias;
    }
}

// ------------------- narrow attn projections (K=256 -> 4) -------------------
__global__ void attn_node_kernel(const float* __restrict__ A,
                                 const float* __restrict__ Was,
                                 const float* __restrict__ Wad,
                                 float* __restrict__ asrc,
                                 float* __restrict__ adst, int Nrows) {
    int t = blockIdx.x * blockDim.x + threadIdx.x;
    int node = t >> 3, slot = t & 7;
    if (node >= Nrows) return;
    int h = slot & 3;
    const float* W = (slot >= 4) ? Wad : Was;
    const float* a = A + (size_t)node * Dk;
    float acc = 0.f;
    for (int k = 0; k < Dk; ++k) acc += a[k] * W[k * NH + h];
    ((slot >= 4) ? adst : asrc)[(size_t)node * NH + h] = acc;
}

// ------------------- edge logits + leaky + segment max ----------------------
__global__ void edge_logits_kernel(const float* __restrict__ fe,
                                   const float* __restrict__ Wae,  // 16x4
                                   const int* __restrict__ src,
                                   const int* __restrict__ dst,
                                   const float* __restrict__ asrc,
                                   const float* __restrict__ adst,
                                   float* __restrict__ e_out,
                                   unsigned* __restrict__ emax, int E) {
    int e = blockIdx.x * blockDim.x + threadIdx.x;
    if (e >= E) return;
    float fx[16];
    const float* row = fe + (size_t)e * 16;
#pragma unroll
    for (int i = 0; i < 16; ++i) fx[i] = row[i];
    int s = src[e], d = dst[e];
#pragma unroll
    for (int h = 0; h < NH; ++h) {
        float acc = 0.f;
#pragma unroll
        for (int i = 0; i < 16; ++i) acc += fx[i] * Wae[i * NH + h];
        float v = leaky(asrc[(size_t)s * NH + h] + adst[(size_t)d * NH + h] + acc);
        e_out[(size_t)e * NH + h] = v;
        atomicMax(&emax[(size_t)d * NH + h], f2ord(v));
    }
}

// ------------------- z = exp(e - max), segment sum --------------------------
__global__ void softmax_z_kernel(const int* __restrict__ dst,
                                 float* __restrict__ ez,          // in-place
                                 const unsigned* __restrict__ emax,
                                 float* __restrict__ zsum, int E) {
    int e = blockIdx.x * blockDim.x + threadIdx.x;
    if (e >= E) return;
    int d = dst[e];
#pragma unroll
    for (int h = 0; h < NH; ++h) {
        float m = ord2f(emax[(size_t)d * NH + h]);
        float z = __expf(ez[(size_t)e * NH + h] - m);
        ez[(size_t)e * NH + h] = z;
        atomicAdd(&zsum[(size_t)d * NH + h], z);
    }
}

// ------------------- msg = fc[src]*a, scatter-add into h --------------------
// 128 threads per edge (coalesced), 2 edges per 256-thread block
__global__ void msg_agg_kernel(const float* __restrict__ fsrc_fc,
                               const float* __restrict__ z,
                               const float* __restrict__ zsum,
                               const int* __restrict__ src,
                               const int* __restrict__ dst,
                               float* __restrict__ h, int E) {
    int lid = threadIdx.x;
    long long e = (long long)blockIdx.x * 2 + (lid >> 7);
    if (e >= E) return;
    int f = lid & (HF - 1);
    int head = f >> 5;
    int s = src[e], d = dst[e];
    float a = z[(size_t)e * NH + head] / zsum[(size_t)d * NH + head];
    atomicAdd(&h[(size_t)d * HF + f], fsrc_fc[(size_t)s * HF + f] * a);
}

// ------------------- per-(node,head) layernorm-affine, in place -------------
__global__ void layernorm_kernel(float* __restrict__ h,
                                 const float* __restrict__ scale,
                                 const float* __restrict__ offset, int Nrows) {
    int gid = blockIdx.x * blockDim.x + threadIdx.x;
    int pair = gid >> 5;                       // one wave32 per (node, head)
    int lane = gid & 31;
    if (pair >= Nrows * NH) return;
    int node = pair >> 2, head = pair & 3;
    size_t idx = (size_t)node * HF + head * Fw + lane;
    float x = h[idx];
    float s = x;
    for (int off = 16; off; off >>= 1) s += __shfl_xor(s, off, 32);
    float mean = s * (1.f / Fw);
    float dx = x - mean;
    float v = dx * dx;
    for (int off = 16; off; off >>= 1) v += __shfl_xor(v, off, 32);
    float var = v * (1.f / Fw) + 1e-9f;
    h[idx] = dx * scale[head * Fw + lane] * rsqrtf(var) + offset[head * Fw + lane];
}

// ------------------- agg GEMM + bias + residual + leaky (in place) ----------
__global__ void agg_gemm_kernel(const float* __restrict__ Hn,
                                const __bf16* __restrict__ WaggSw,
                                const float* __restrict__ b_agg,
                                float* __restrict__ RstIO, int Nrows) {
    __shared__ __align__(32) uint32_t As[8 * HF];   // 16 x 128 bf16, swizzled
    const int row0 = blockIdx.x * 16;
    stage_A_sw<HF>(As, Hn, row0, Nrows);
    __syncthreads();
    const int ct = threadIdx.x >> 5;
    v8f c = wmma_tile_sw<HF>(As, WaggSw + (size_t)ct * HF * 16);
    const int lane = threadIdx.x & 31;
    const int hh = lane >> 4, nn = lane & 15;
    const int col0 = ct << 4;
    const float bias = b_agg[col0 + nn];
#pragma unroll
    for (int v = 0; v < 8; ++v) {
        int row = row0 + v + (hh << 3);
        if (row < Nrows) {
            size_t idx = (size_t)row * HF + col0 + nn;
            RstIO[idx] = leaky(c[v] + bias + RstIO[idx]);
        }
    }
}

// ------------------- final GEMM: rst @ W_apply + b -> out -------------------
// block = 256 (8 waves): 4 row tiles x 2 col tiles = 64 rows x 32 cols
__global__ void apply_gemm_kernel(const float* __restrict__ Rst,
                                  const __bf16* __restrict__ WapSw,
                                  const float* __restrict__ b_ap,
                                  float* __restrict__ Out, int Nrows) {
    __shared__ __align__(32) uint32_t As[4][8 * HF];   // 4 x (16x128) swizzled
    const int rowBase = blockIdx.x * 64;
#pragma unroll
    for (int rt = 0; rt < 4; ++rt)
        stage_A_sw<HF>(As[rt], Rst, rowBase + rt * 16, Nrows);
    __syncthreads();
    const int w = threadIdx.x >> 5;
    const int rt = w >> 1, ct = w & 1;
    const int row0 = rowBase + rt * 16;
    const int col0 = ct << 4;
    v8f c = wmma_tile_sw<HF>(As[rt], WapSw + (size_t)ct * HF * 16);
    const int lane = threadIdx.x & 31;
    const int hh = lane >> 4, nn = lane & 15;
    const float bias = b_ap[col0 + nn];
#pragma unroll
    for (int v = 0; v < 8; ++v) {
        int row = row0 + v + (hh << 3);
        if (row < Nrows) Out[(size_t)row * OUTW + col0 + nn] = c[v] + bias;
    }
}

// ---------------------------------------------------------------------------
extern "C" void kernel_launch(void* const* d_in, const int* in_sizes, int n_in,
                              void* d_out, int out_size, void* d_ws, size_t ws_size,
                              hipStream_t stream) {
    (void)n_in; (void)out_size; (void)ws_size;
    const float* feat_src   = (const float*)d_in[0];
    const float* feat_edge  = (const float*)d_in[1];
    const int*   src_idx    = (const int*)d_in[2];
    const int*   dst_idx    = (const int*)d_in[3];
    const float* W_src      = (const float*)d_in[4];
    const float* W_attn_src = (const float*)d_in[5];
    const float* W_attn_dst = (const float*)d_in[6];
    const float* W_attn_edge= (const float*)d_in[7];
    const float* scale      = (const float*)d_in[8];
    const float* offset     = (const float*)d_in[9];
    const float* W_agg      = (const float*)d_in[10];
    const float* b_agg      = (const float*)d_in[11];
    const float* W_dst      = (const float*)d_in[12];
    const float* b_dst      = (const float*)d_in[13];
    const float* W_apply    = (const float*)d_in[14];
    const float* b_apply    = (const float*)d_in[15];

    const int N = in_sizes[0] / Dk;
    const int E = in_sizes[2];

    // ---- carve scratch ----
    char* p = (char*)d_ws;
    auto carve = [&](size_t bytes) {
        void* r = (void*)p;
        p += (bytes + 255) & ~(size_t)255;
        return r;
    };
    float*    ws_fsrc  = (float*)carve((size_t)N * HF * 4);   // feat_src_fc
    float*    ws_fdst  = (float*)carve((size_t)N * HF * 4);   // feat_dst_fc / rst
    float*    ws_h     = (float*)carve((size_t)N * HF * 4);   // aggregated h
    float*    ws_asrc  = (float*)carve((size_t)N * NH * 4);
    float*    ws_adst  = (float*)carve((size_t)N * NH * 4);
    unsigned* ws_emax  = (unsigned*)carve((size_t)N * NH * 4);
    float*    ws_zsum  = (float*)carve((size_t)N * NH * 4);
    float*    ws_e     = (float*)carve((size_t)E * NH * 4);   // logits -> z
    uint32_t* wb_src   = (uint32_t*)carve((size_t)Dk * HF * 2);
    uint32_t* wb_dst   = (uint32_t*)carve((size_t)Dk * HF * 2);
    uint32_t* wb_agg   = (uint32_t*)carve((size_t)HF * HF * 2);
    uint32_t* wb_apply = (uint32_t*)carve((size_t)HF * OUTW * 2);

    // 1) weights -> bf16, pre-swizzled into WMMA B-operand layout
    cvt_swz_kernel<<<(Dk * HF / 2 + 255) / 256, 256, 0, stream>>>(W_src, wb_src, Dk, HF);
    cvt_swz_kernel<<<(Dk * HF / 2 + 255) / 256, 256, 0, stream>>>(W_dst, wb_dst, Dk, HF);
    cvt_swz_kernel<<<(HF * HF / 2 + 255) / 256, 256, 0, stream>>>(W_agg, wb_agg, HF, HF);
    cvt_swz_kernel<<<(HF * OUTW / 2 + 255) / 256, 256, 0, stream>>>(W_apply, wb_apply, HF, OUTW);

    // 2) init accumulators
    init_kernel<<<(N * HF + 255) / 256, 256, 0, stream>>>(ws_h, ws_zsum, ws_emax,
                                                          N * HF, N * NH);

    // 3) WMMA node projections (src + dst, fused bias)
    node_proj_kernel<<<(N + 15) / 16, 512, 0, stream>>>(feat_src, (const __bf16*)wb_src,
                                                        (const __bf16*)wb_dst, b_dst,
                                                        ws_fsrc, ws_fdst, N);

    // 4) attention projections (narrow)
    attn_node_kernel<<<(N * 8 + 255) / 256, 256, 0, stream>>>(feat_src, W_attn_src,
                                                              W_attn_dst, ws_asrc,
                                                              ws_adst, N);

    // 5) edge logits + segment max
    edge_logits_kernel<<<(E + 255) / 256, 256, 0, stream>>>(feat_edge, W_attn_edge,
                                                            src_idx, dst_idx, ws_asrc,
                                                            ws_adst, ws_e, ws_emax, E);

    // 6) z = exp(e - max), segment sum
    softmax_z_kernel<<<(E + 255) / 256, 256, 0, stream>>>(dst_idx, ws_e, ws_emax,
                                                          ws_zsum, E);

    // 7) message gather/scatter-add
    msg_agg_kernel<<<(E + 1) / 2, 256, 0, stream>>>(ws_fsrc, ws_e, ws_zsum, src_idx,
                                                    dst_idx, ws_h, E);

    // 8) layernorm-affine (in place)
    layernorm_kernel<<<(N * NH + 7) / 8, 256, 0, stream>>>(ws_h, scale, offset, N);

    // 9) WMMA agg GEMM + bias + residual + leaky (into ws_fdst)
    agg_gemm_kernel<<<(N + 15) / 16, 256, 0, stream>>>(ws_h, (const __bf16*)wb_agg,
                                                       b_agg, ws_fdst, N);

    // 10) WMMA apply GEMM + bias -> out
    apply_gemm_kernel<<<(N + 63) / 64, 256, 0, stream>>>(ws_fdst, (const __bf16*)wb_apply,
                                                         b_apply, (float*)d_out, N);
}
